// GAT_44813688766474
// MI455X (gfx1250) — compile-verified, matrix operand order checked by reading, not run on hardware
//
#include <hip/hip_runtime.h>
#include <hip/hip_bf16.h>
#include <math.h>

// ---------------------------------------------------------------------------
// Types for gfx1250 WMMA (wave32)
// ---------------------------------------------------------------------------
typedef __bf16 bf16_t;
typedef __attribute__((ext_vector_type(16))) __bf16 v16bf;
typedef __attribute__((ext_vector_type(8)))  __bf16 v8bf;
typedef __attribute__((ext_vector_type(8)))  float  v8f;

__device__ __forceinline__ bf16_t f32_to_bf16(float f) {
  unsigned u = __float_as_uint(f);
  u += 0x7FFFu + ((u >> 16) & 1u);          // round-to-nearest-even
  unsigned short h = (unsigned short)(u >> 16);
  return __builtin_bit_cast(bf16_t, h);
}

__device__ __forceinline__ float wave_reduce_sum(float v) {
  #pragma unroll
  for (int off = 16; off > 0; off >>= 1) v += __shfl_xor(v, off, 32);
  return v;
}

// Order-preserving float atomic max (init must be -inf)
__device__ __forceinline__ void atomicMaxFloat(float* addr, float val) {
  if (val >= 0.0f) atomicMax((int*)addr, __float_as_int(val));
  else             atomicMin((unsigned int*)addr, __float_as_uint(val));
}

// ---------------------------------------------------------------------------
// Small utility kernels
// ---------------------------------------------------------------------------
__global__ void fill_kernel(float* p, float v, int n) {
  int t = blockIdx.x * blockDim.x + threadIdx.x;
  if (t < n) p[t] = v;
}

__global__ void cvt_bf16_kernel(const float* __restrict__ x, bf16_t* __restrict__ y, int n) {
  int t = blockIdx.x * blockDim.x + threadIdx.x;
  if (t < n) y[t] = f32_to_bf16(x[t]);
}

// W [K, Nc] fp32 row-major -> Wt [Nc, K] bf16 row-major (i.e. column-major W)
__global__ void transpose_cvt_kernel(const float* __restrict__ W, bf16_t* __restrict__ Wt,
                                     int K, int Nc) {
  int t = blockIdx.x * blockDim.x + threadIdx.x;
  if (t >= K * Nc) return;
  int k = t / Nc, n1 = t % Nc;
  Wt[(size_t)n1 * K + k] = f32_to_bf16(W[t]);
}

// vd[h*K + k] = sum_c W[k, h*C + c] * att[h, c]   (folds lin_dst GEMM into a GEMV)
__global__ void fold_att_kernel(const float* __restrict__ W, const float* __restrict__ att,
                                float* __restrict__ vd, int K, int C, int H, int Nc) {
  int t = blockIdx.x * blockDim.x + threadIdx.x;
  if (t >= K * H) return;
  int h = t / K, k = t % K;
  const float* wrow = W + (size_t)k * Nc + (size_t)h * C;
  const float* a    = att + (size_t)h * C;
  float s = 0.f;
  for (int c = 0; c < C; ++c) s += wrow[c] * a[c];
  vd[t] = s;
}

__global__ void build_edges_kernel(const int* __restrict__ ei, int* __restrict__ src,
                                   int* __restrict__ dst, int E, int N) {
  int t = blockIdx.x * blockDim.x + threadIdx.x;
  if (t < E)          { src[t] = ei[t]; dst[t] = ei[E + t]; }
  else if (t < E + N) { src[t] = t - E; dst[t] = t - E; }     // self-loops
}

// ---------------------------------------------------------------------------
// bf16 WMMA GEMM: C[M,Nc] = A[M,K] @ B[K,Nc] (+bias)(+relu)
// A row-major bf16, Bt = B^T row-major bf16 ([Nc,K]). One wave per 16x16 tile.
// Fragment layout per ISA: 16-bit A/B, lane<16 -> K {0..7,16..23}, lane>=16 -> K {8..15,24..31}.
// ---------------------------------------------------------------------------
__global__ void gemm_bf16_wmma_kernel(const bf16_t* __restrict__ A, const bf16_t* __restrict__ Bt,
                                      float* __restrict__ C, const float* __restrict__ bias,
                                      int M, int K, int Nc, int relu) {
  int lane = threadIdx.x & 31;
  int wave = threadIdx.x >> 5;
  int nt = blockIdx.x;
  int mt = blockIdx.y * 4 + wave;
  if (mt * 16 >= M) return;

  int row   = mt * 16 + (lane & 15);
  int col   = nt * 16 + (lane & 15);
  int khalf = (lane >> 4) << 3;           // 0 or 8

  const bf16_t* arow = A  + (size_t)row * K;
  const bf16_t* brow = Bt + (size_t)col * K;

  v8f acc = {0.f, 0.f, 0.f, 0.f, 0.f, 0.f, 0.f, 0.f};

  for (int k0 = 0; k0 < K; k0 += 32) {
    v8bf alo = *(const v8bf*)(arow + k0 + khalf);
    v8bf ahi = *(const v8bf*)(arow + k0 + 16 + khalf);
    v8bf blo = *(const v8bf*)(brow + k0 + khalf);
    v8bf bhi = *(const v8bf*)(brow + k0 + 16 + khalf);
    v16bf a = __builtin_shufflevector(alo, ahi, 0,1,2,3,4,5,6,7,8,9,10,11,12,13,14,15);
    v16bf b = __builtin_shufflevector(blo, bhi, 0,1,2,3,4,5,6,7,8,9,10,11,12,13,14,15);
    acc = __builtin_amdgcn_wmma_f32_16x16x32_bf16(false, a, false, b, (short)0, acc,
                                                  false, false);
  }

  int rbase = mt * 16 + ((lane >> 4) << 3);   // C layout: VGPR r -> M = r (+8 for upper lanes)
  float bv = bias ? bias[col] : 0.f;
  #pragma unroll
  for (int r = 0; r < 8; ++r) {
    float v = acc[r] + bv;
    if (relu) v = v > 0.f ? v : 0.f;
    C[(size_t)(rbase + r) * Nc + col] = v;
  }
}

// ---------------------------------------------------------------------------
// out[i*H+h] = dot(X[i, (perHead? h*len : 0) .. +len], V[h, 0..len])  (wave per (i,h))
// ---------------------------------------------------------------------------
__global__ void rowdot_kernel(const float* __restrict__ X, const float* __restrict__ V,
                              float* __restrict__ out, int n, int H, int len,
                              int xstride, int perHead) {
  int gw   = (blockIdx.x * blockDim.x + threadIdx.x) >> 5;
  int lane = threadIdx.x & 31;
  if (gw >= n * H) return;
  int i = gw / H, h = gw % H;
  const float* xr = X + (size_t)i * xstride + (perHead ? (size_t)h * len : 0);
  const float* vr = V + (size_t)h * len;
  float s = 0.f;
  for (int c = lane; c < len; c += 32) s += xr[c] * vr[c];
  s = wave_reduce_sum(s);
  if (lane == 0) out[gw] = s;
}

// ---------------------------------------------------------------------------
// Edge softmax passes
// ---------------------------------------------------------------------------
__global__ void edge_pass1_kernel(const int* __restrict__ src, const int* __restrict__ dst,
                                  const float* __restrict__ as_, const float* __restrict__ ad_,
                                  float* __restrict__ alpha, float* __restrict__ m,
                                  int Ecnt, int H) {
  int t = blockIdx.x * blockDim.x + threadIdx.x;
  if (t >= Ecnt * H) return;
  int e = t / H, h = t % H;
  int s = src[e], d = dst[e];
  float v = as_[s * H + h] + ad_[d * H + h];
  v = v > 0.f ? v : 0.2f * v;                 // leaky_relu(0.2)
  alpha[t] = v;
  atomicMaxFloat(&m[d * H + h], v);
}

__global__ void edge_pass2_kernel(const int* __restrict__ dst, float* __restrict__ alpha,
                                  const float* __restrict__ m, float* __restrict__ den,
                                  int Ecnt, int H) {
  int t = blockIdx.x * blockDim.x + threadIdx.x;
  if (t >= Ecnt * H) return;
  int e = t / H, h = t % H;
  int d = dst[e];
  float v = __expf(alpha[t] - m[d * H + h]);
  alpha[t] = v;
  atomicAdd(&den[d * H + h], v);
}

// out[dst, h, :] += alpha/den * xs[src, h, :]   (wave per (edge, head))
__global__ void aggregate_kernel(const int* __restrict__ src, const int* __restrict__ dst,
                                 const float* __restrict__ xs, const float* __restrict__ alpha,
                                 const float* __restrict__ den, float* __restrict__ out,
                                 int Ecnt, int H, int C) {
  int gw   = (blockIdx.x * blockDim.x + threadIdx.x) >> 5;
  int lane = threadIdx.x & 31;
  if (gw >= Ecnt * H) return;
  int e = gw / H, h = gw % H;
  int s = src[e], d = dst[e];
  float w = alpha[gw] / den[d * H + h];
  size_t sb = (size_t)s * H * C + (size_t)h * C;
  size_t db = (size_t)d * H * C + (size_t)h * C;
  for (int c = lane; c < C; c += 32)
    atomicAdd(&out[db + c], xs[sb + c] * w);
}

// io = relu(io + b[col]);  ob = bf16(io)
__global__ void bias_relu_cvt_kernel(float* __restrict__ io, const float* __restrict__ b,
                                     bf16_t* __restrict__ ob, int n, int Nc) {
  int t = blockIdx.x * blockDim.x + threadIdx.x;
  if (t >= n) return;
  float v = io[t] + b[t % Nc];
  v = v > 0.f ? v : 0.f;
  io[t] = v;
  ob[t] = f32_to_bf16(v);
}

// ---------------------------------------------------------------------------
// BatchNorm stats: one block per feature; stats = [mean(F), rstd(F)]
// ---------------------------------------------------------------------------
__global__ void bn_stats_kernel(const float* __restrict__ hf, float* __restrict__ stats,
                                int n, int F) {
  __shared__ float ss[256], sq[256];
  int f = blockIdx.x;
  float s = 0.f, q = 0.f;
  for (int i = threadIdx.x; i < n; i += blockDim.x) {
    float v = hf[(size_t)i * F + f];
    s += v; q += v * v;
  }
  ss[threadIdx.x] = s; sq[threadIdx.x] = q;
  __syncthreads();
  for (int off = 128; off > 0; off >>= 1) {
    if ((int)threadIdx.x < off) { ss[threadIdx.x] += ss[threadIdx.x + off];
                                  sq[threadIdx.x] += sq[threadIdx.x + off]; }
    __syncthreads();
  }
  if (threadIdx.x == 0) {
    float mean = ss[0] / n;
    float var  = sq[0] / n - mean * mean;
    stats[f]     = mean;
    stats[F + f] = rsqrtf(var + 1e-5f);
  }
}

// BN affine + relu + [128 x 3] matvec + log_softmax, one wave per node
__global__ void final_kernel(const float* __restrict__ hf, const float* __restrict__ stats,
                             const float* __restrict__ gamma, const float* __restrict__ beta,
                             const float* __restrict__ wf2, const float* __restrict__ bf2,
                             float* __restrict__ out, int n, int F) {
  int gw   = (blockIdx.x * blockDim.x + threadIdx.x) >> 5;
  int lane = threadIdx.x & 31;
  if (gw >= n) return;
  float a0 = 0.f, a1 = 0.f, a2 = 0.f;
  for (int f = lane; f < F; f += 32) {
    float v = (hf[(size_t)gw * F + f] - stats[f]) * stats[F + f] * gamma[f] + beta[f];
    v = v > 0.f ? v : 0.f;
    a0 += v * wf2[f * 3 + 0];
    a1 += v * wf2[f * 3 + 1];
    a2 += v * wf2[f * 3 + 2];
  }
  a0 = wave_reduce_sum(a0);
  a1 = wave_reduce_sum(a1);
  a2 = wave_reduce_sum(a2);
  if (lane == 0) {
    float y0 = a0 + bf2[0], y1 = a1 + bf2[1], y2 = a2 + bf2[2];
    float mx  = fmaxf(y0, fmaxf(y1, y2));
    float lse = mx + logf(__expf(y0 - mx) + __expf(y1 - mx) + __expf(y2 - mx));
    out[gw * 3 + 0] = y0 - lse;
    out[gw * 3 + 1] = y1 - lse;
    out[gw * 3 + 2] = y2 - lse;
  }
}

// ---------------------------------------------------------------------------
// Host orchestration
// ---------------------------------------------------------------------------
static inline int cdiv(int a, int b) { return (a + b - 1) / b; }

extern "C" void kernel_launch(void* const* d_in, const int* in_sizes, int n_in,
                              void* d_out, int out_size, void* d_ws, size_t ws_size,
                              hipStream_t stream) {
  const int DIN = 256, H = 2, C1 = 256, C2 = 512, F = 128;
  const int HC1 = H * C1;     // 512
  const int HC2 = H * C2;     // 1024
  const int N    = in_sizes[0] / DIN;   // 10000
  const int E    = in_sizes[1] / 2;     // 160000
  const int Ecnt = E + N;               // + self-loops

  const float* x   = (const float*)d_in[0];
  const int*   ei  = (const int*)  d_in[1];
  const float* w1s = (const float*)d_in[2];
  const float* w1d = (const float*)d_in[3];
  const float* a1s = (const float*)d_in[4];
  const float* a1d = (const float*)d_in[5];
  const float* b1  = (const float*)d_in[6];
  const float* w2s = (const float*)d_in[7];
  const float* w2d = (const float*)d_in[8];
  const float* a2s = (const float*)d_in[9];
  const float* a2d = (const float*)d_in[10];
  const float* b2  = (const float*)d_in[11];
  const float* wf1 = (const float*)d_in[12];
  const float* bf1 = (const float*)d_in[13];
  const float* gmm = (const float*)d_in[14];
  const float* bta = (const float*)d_in[15];
  const float* wf2 = (const float*)d_in[16];
  const float* bf2 = (const float*)d_in[17];
  float* out = (float*)d_out;

  // --- workspace arena ---
  char* base = (char*)d_ws;
  size_t off = 0;
  auto carve = [&](size_t bytes) -> void* {
    void* p = base + off;
    off = (off + bytes + 255) & ~(size_t)255;
    return p;
  };
  int*    src   = (int*)   carve((size_t)Ecnt * 4);
  int*    dst   = (int*)   carve((size_t)Ecnt * 4);
  bf16_t* xb    = (bf16_t*)carve((size_t)N * DIN * 2);
  bf16_t* w1sT  = (bf16_t*)carve((size_t)DIN * HC1 * 2);
  bf16_t* w2sT  = (bf16_t*)carve((size_t)HC1 * HC2 * 2);
  bf16_t* wf1T  = (bf16_t*)carve((size_t)HC2 * F * 2);
  float*  vd1   = (float*) carve((size_t)H * DIN * 4);
  float*  vd2   = (float*) carve((size_t)H * HC1 * 4);
  float*  xs1   = (float*) carve((size_t)N * HC1 * 4);
  float*  xs2   = (float*) carve((size_t)N * HC2 * 4);
  float*  out1  = (float*) carve((size_t)N * HC1 * 4);
  float*  out2  = (float*) carve((size_t)N * HC2 * 4);
  bf16_t* h1b   = (bf16_t*)carve((size_t)N * HC1 * 2);
  bf16_t* h2b   = (bf16_t*)carve((size_t)N * HC2 * 2);
  float*  as_   = (float*) carve((size_t)N * H * 4);
  float*  ad_   = (float*) carve((size_t)N * H * 4);
  float*  m_    = (float*) carve((size_t)N * H * 4);
  float*  den_  = (float*) carve((size_t)N * H * 4);
  float*  alpha = (float*) carve((size_t)Ecnt * H * 4);
  float*  hf    = (float*) carve((size_t)N * F * 4);
  float*  stats = (float*) carve((size_t)2 * F * 4);
  (void)ws_size; (void)n_in; (void)out_size;

  const int B = 256;
  const float NEG_INF = -__builtin_inff();

  // --- preprocessing ---
  build_edges_kernel<<<cdiv(Ecnt, B), B, 0, stream>>>(ei, src, dst, E, N);
  cvt_bf16_kernel<<<cdiv(N * DIN, B), B, 0, stream>>>(x, xb, N * DIN);
  transpose_cvt_kernel<<<cdiv(DIN * HC1, B), B, 0, stream>>>(w1s, w1sT, DIN, HC1);
  transpose_cvt_kernel<<<cdiv(HC1 * HC2, B), B, 0, stream>>>(w2s, w2sT, HC1, HC2);
  transpose_cvt_kernel<<<cdiv(HC2 * F,  B), B, 0, stream>>>(wf1, wf1T, HC2, F);
  fold_att_kernel<<<cdiv(DIN * H, B), B, 0, stream>>>(w1d, a1d, vd1, DIN, C1, H, HC1);
  fold_att_kernel<<<cdiv(HC1 * H, B), B, 0, stream>>>(w2d, a2d, vd2, HC1, C2, H, HC2);

  // ================== GAT layer 1 ==================
  gemm_bf16_wmma_kernel<<<dim3(HC1 / 16, cdiv(N / 16, 4)), 128, 0, stream>>>(
      xb, w1sT, xs1, nullptr, N, DIN, HC1, 0);
  rowdot_kernel<<<cdiv(N * H * 32, B), B, 0, stream>>>(xs1, a1s, as_, N, H, C1, HC1, 1);
  rowdot_kernel<<<cdiv(N * H * 32, B), B, 0, stream>>>(x,   vd1, ad_, N, H, DIN, DIN, 0);

  fill_kernel<<<cdiv(N * H, B), B, 0, stream>>>(m_,   NEG_INF, N * H);
  fill_kernel<<<cdiv(N * H, B), B, 0, stream>>>(den_, 0.f,     N * H);
  fill_kernel<<<cdiv(N * HC1, B), B, 0, stream>>>(out1, 0.f,   N * HC1);

  edge_pass1_kernel<<<cdiv(Ecnt * H, B), B, 0, stream>>>(src, dst, as_, ad_, alpha, m_, Ecnt, H);
  edge_pass2_kernel<<<cdiv(Ecnt * H, B), B, 0, stream>>>(dst, alpha, m_, den_, Ecnt, H);
  aggregate_kernel<<<cdiv(Ecnt * H * 32, B), B, 0, stream>>>(src, dst, xs1, alpha, den_, out1,
                                                             Ecnt, H, C1);
  bias_relu_cvt_kernel<<<cdiv(N * HC1, B), B, 0, stream>>>(out1, b1, h1b, N * HC1, HC1);

  // ================== GAT layer 2 ==================
  gemm_bf16_wmma_kernel<<<dim3(HC2 / 16, cdiv(N / 16, 4)), 128, 0, stream>>>(
      h1b, w2sT, xs2, nullptr, N, HC1, HC2, 0);
  rowdot_kernel<<<cdiv(N * H * 32, B), B, 0, stream>>>(xs2, a2s, as_, N, H, C2, HC2, 1);
  rowdot_kernel<<<cdiv(N * H * 32, B), B, 0, stream>>>(out1, vd2, ad_, N, H, HC1, HC1, 0);

  fill_kernel<<<cdiv(N * H, B), B, 0, stream>>>(m_,   NEG_INF, N * H);
  fill_kernel<<<cdiv(N * H, B), B, 0, stream>>>(den_, 0.f,     N * H);
  fill_kernel<<<cdiv(N * HC2, B), B, 0, stream>>>(out2, 0.f,   N * HC2);

  edge_pass1_kernel<<<cdiv(Ecnt * H, B), B, 0, stream>>>(src, dst, as_, ad_, alpha, m_, Ecnt, H);
  edge_pass2_kernel<<<cdiv(Ecnt * H, B), B, 0, stream>>>(dst, alpha, m_, den_, Ecnt, H);
  aggregate_kernel<<<cdiv(Ecnt * H * 32, B), B, 0, stream>>>(src, dst, xs2, alpha, den_, out2,
                                                             Ecnt, H, C2);
  bias_relu_cvt_kernel<<<cdiv(N * HC2, B), B, 0, stream>>>(out2, b2, h2b, N * HC2, HC2);

  // ================== MLP head ==================
  gemm_bf16_wmma_kernel<<<dim3(F / 16, cdiv(N / 16, 4)), 128, 0, stream>>>(
      h2b, wf1T, hf, bf1, N, HC2, F, 0);
  bn_stats_kernel<<<F, 256, 0, stream>>>(hf, stats, N, F);
  final_kernel<<<cdiv(N * 32, B), B, 0, stream>>>(hf, stats, gmm, bta, wf2, bf2, out, N, F);
}
